// HandGestureNet_9887014716175
// MI455X (gfx1250) — compile-verified, matrix operand order checked by reading, not run on hardware
//
#include <hip/hip_runtime.h>
#include <hip/hip_bf16.h>

typedef __attribute__((ext_vector_type(16))) _Float16 v16h;
typedef __attribute__((ext_vector_type(8)))  _Float16 v8h;
typedef __attribute__((ext_vector_type(8)))  float    v8f;

#define NB     4096
#define NC     66
#define LSEQ   100
#define NFEAT  7128     // 9 * 66 * 12
#define NFEATP 7136     // padded to multiple of 32
#define NH     1936
#define NHP    1952     // padded to multiple of 32
#define NOUT   14

// ---------------- weight conversion with zero K-padding ----------------
__global__ void cvt_w1(const float* __restrict__ src, _Float16* __restrict__ dst) {
  int i = blockIdx.x * blockDim.x + threadIdx.x;
  if (i >= NH * NFEATP) return;
  int row = i / NFEATP, k = i % NFEATP;
  dst[i] = (k < NFEAT) ? (_Float16)src[row * NFEAT + k] : (_Float16)0.f;
}

__global__ void cvt_w2(const float* __restrict__ src, _Float16* __restrict__ dst) {
  int i = blockIdx.x * blockDim.x + threadIdx.x;
  if (i >= 16 * NHP) return;
  int row = i / NHP, k = i % NHP;
  dst[i] = (row < NOUT && k < NH) ? (_Float16)src[row * NH + k] : (_Float16)0.f;
}

__global__ void zero_hpad(_Float16* __restrict__ H) {
  int i = blockIdx.x * blockDim.x + threadIdx.x;
  if (i >= NB * 16) return;
  H[(size_t)(i >> 4) * NHP + NH + (i & 15)] = (_Float16)0.f;
}

// ---------------- per-(b,c) conv feature extractor ----------------
struct WaveBuf {
  float xp[108];          // 3-pad + 100 + 3-pad
  float p1[2][8][56];     // [branch][oc][3 + 50 + 3]
  float p2[2][4][32];     // [branch][oc][3 + 25 + pad]
  float w1[2][8][8];
  float w2[2][4][8][8];
  float w3[2][4][4][8];
  float b1[2][8];
  float b2[2][4];
  float b3[2][4];
};

__global__ __launch_bounds__(128) void features_kernel(
    const float* __restrict__ x,
    const float* __restrict__ hw1, const float* __restrict__ hb1,
    const float* __restrict__ hw2, const float* __restrict__ hb2,
    const float* __restrict__ hw3, const float* __restrict__ hb3,
    const float* __restrict__ lw1, const float* __restrict__ lb1,
    const float* __restrict__ lw2, const float* __restrict__ lb2,
    const float* __restrict__ lw3, const float* __restrict__ lb3,
    _Float16* __restrict__ F) {
  __shared__ WaveBuf sm[4];
  const int wave = threadIdx.x >> 5;
  const int lane = threadIdx.x & 31;
  const int pair = blockIdx.x * 4 + wave;
  const int b = pair / NC;
  const int c = pair % NC;
  WaveBuf& W = sm[wave];

  {
    float* p = (float*)&W;
    const int n = (int)(sizeof(WaveBuf) / 4);
    for (int i = lane; i < n; i += 32) p[i] = 0.0f;
  }
  __syncthreads();

  // weights (low branch shifted by +2 into a 7-tap frame)
  for (int i = lane; i < 56;  i += 32) { int oc=i/7,  k=i%7;                 W.w1[0][oc][k]       = hw1[c*56  + i]; }
  for (int i = lane; i < 24;  i += 32) { int oc=i/3,  k=i%3;                 W.w1[1][oc][2+k]     = lw1[c*24  + i]; }
  for (int i = lane; i < 224; i += 32) { int oc=i/56, r=i%56, ic=r/7, k=r%7; W.w2[0][oc][ic][k]   = hw2[c*224 + i]; }
  for (int i = lane; i < 96;  i += 32) { int oc=i/24, r=i%24, ic=r/3, k=r%3; W.w2[1][oc][ic][2+k] = lw2[c*96  + i]; }
  for (int i = lane; i < 112; i += 32) { int oc=i/28, r=i%28, ic=r/7, k=r%7; W.w3[0][oc][ic][k]   = hw3[c*112 + i]; }
  for (int i = lane; i < 48;  i += 32) { int oc=i/12, r=i%12, ic=r/3, k=r%3; W.w3[1][oc][ic][2+k] = lw3[c*48  + i]; }
  if (lane < 8) { W.b1[0][lane] = hb1[c*8+lane]; W.b1[1][lane] = lb1[c*8+lane]; }
  if (lane < 4) {
    W.b2[0][lane] = hb2[c*4+lane]; W.b2[1][lane] = lb2[c*4+lane];
    W.b3[0][lane] = hb3[c*4+lane]; W.b3[1][lane] = lb3[c*4+lane];
  }
  for (int l = lane; l < LSEQ; l += 32)
    W.xp[3 + l] = x[(size_t)b * (LSEQ * NC) + (size_t)l * NC + c];
  // zero the K-pad of this batch row (once, by the c==0 wave)
  if (c == 0 && lane < (NFEATP - NFEAT))
    F[(size_t)b * NFEATP + NFEAT + lane] = (_Float16)0.f;
  __syncthreads();

  // stage A: conv1 + relu + pool2 -> p1[br][8][50]
  for (int i = lane; i < 800; i += 32) {
    int br = i / 400, r = i % 400, oc = r / 50, t = r % 50;
    float s0 = W.b1[br][oc], s1 = s0;
    #pragma unroll
    for (int k = 0; k < 7; ++k) {
      float wk = W.w1[br][oc][k];
      s0 += W.xp[2*t + k]     * wk;
      s1 += W.xp[2*t + 1 + k] * wk;
    }
    s0 = s0 > 0.f ? s0 : 0.f;
    s1 = s1 > 0.f ? s1 : 0.f;
    W.p1[br][oc][3 + t] = 0.5f * (s0 + s1);
  }
  __syncthreads();

  // stage B: conv2 + relu + pool2 -> p2[br][4][25]
  for (int i = lane; i < 200; i += 32) {
    int br = i / 100, r = i % 100, oc = r / 25, t = r % 25;
    float s0 = W.b2[br][oc], s1 = s0;
    for (int ic = 0; ic < 8; ++ic) {
      #pragma unroll
      for (int k = 0; k < 7; ++k) {
        float wk = W.w2[br][oc][ic][k];
        s0 += W.p1[br][ic][2*t + k]     * wk;
        s1 += W.p1[br][ic][2*t + 1 + k] * wk;
      }
    }
    s0 = s0 > 0.f ? s0 : 0.f;
    s1 = s1 > 0.f ? s1 : 0.f;
    W.p2[br][oc][3 + t] = 0.5f * (s0 + s1);
  }
  __syncthreads();

  // stage C: conv3 + relu + pool2 -> feature channels 0..7
  _Float16* Fbc = F + (size_t)b * NFEATP + c * 108;
  for (int i = lane; i < 96; i += 32) {
    int br = i / 48, r = i % 48, oc = r / 12, t = r % 12;
    float s0 = W.b3[br][oc], s1 = s0;
    for (int ic = 0; ic < 4; ++ic) {
      #pragma unroll
      for (int k = 0; k < 7; ++k) {
        float wk = W.w3[br][oc][ic][k];
        s0 += W.p2[br][ic][2*t + k]     * wk;
        s1 += W.p2[br][ic][2*t + 1 + k] * wk;
      }
    }
    s0 = s0 > 0.f ? s0 : 0.f;
    s1 = s1 > 0.f ? s1 : 0.f;
    Fbc[(br * 4 + oc) * 12 + t] = (_Float16)(0.5f * (s0 + s1));
  }
  // stage D: triple-pooled residual of x -> feature channel 8
  if (lane < 12) {
    float s = 0.f;
    #pragma unroll
    for (int j = 0; j < 8; ++j) s += W.xp[3 + 8 * lane + j];
    Fbc[96 + lane] = (_Float16)(s * 0.125f);
  }
}

// ---------------- GEMM1: H = relu(F @ W1^T + fb1), f16 out ----------------
// 4x4 register tile per wave (64 rows x 64 cols): 16 WMMAs per 8 operand
// fetches; K padded so the inner loop has no conditionals.
__global__ __launch_bounds__(256) void gemm1_kernel(
    const _Float16* __restrict__ F, const _Float16* __restrict__ W1,
    const float* __restrict__ fb1, _Float16* __restrict__ H) {
  const int wave = threadIdx.x >> 5, lane = threadIdx.x & 31;
  const int r16 = lane & 15, hi = lane >> 4;
  const int mbase = (blockIdx.y * 8 + wave) * 64;
  const int nbase = blockIdx.x * 64;

  const _Float16* abase = F + (size_t)(mbase + r16) * NFEATP + hi * 8;
  const _Float16* brow[4];
  #pragma unroll
  for (int nj = 0; nj < 4; ++nj) {
    int nr = nbase + nj * 16 + r16;
    if (nr > NH - 1) nr = NH - 1;       // clamp: last N-group only
    brow[nj] = W1 + (size_t)nr * NFEATP + hi * 8;
  }

  v8f acc[4][4] = {};
  for (int k0 = 0; k0 < NFEATP; k0 += 32) {
    v16h a[4], b[4];
    #pragma unroll
    for (int mi = 0; mi < 4; ++mi) {
      const _Float16* p = abase + (size_t)mi * 16 * NFEATP + k0;
      v8h lo = *(const v8h*)(p);
      v8h hh = *(const v8h*)(p + 16);
      a[mi] = __builtin_shufflevector(lo, hh, 0,1,2,3,4,5,6,7,8,9,10,11,12,13,14,15);
    }
    #pragma unroll
    for (int nj = 0; nj < 4; ++nj) {
      const _Float16* p = brow[nj] + k0;
      v8h lo = *(const v8h*)(p);
      v8h hh = *(const v8h*)(p + 16);
      b[nj] = __builtin_shufflevector(lo, hh, 0,1,2,3,4,5,6,7,8,9,10,11,12,13,14,15);
    }
    #pragma unroll
    for (int mi = 0; mi < 4; ++mi)
      #pragma unroll
      for (int nj = 0; nj < 4; ++nj)
        acc[mi][nj] = __builtin_amdgcn_wmma_f32_16x16x32_f16(
            false, a[mi], false, b[nj], (short)0, acc[mi][nj], false, false);
    // stream-ahead prefetch (global_prefetch_b8), ~2KB ahead per row
    #pragma unroll
    for (int mi = 0; mi < 4; ++mi)
      __builtin_prefetch(abase + (size_t)mi * 16 * NFEATP + k0 + 1024, 0, 1);
    #pragma unroll
    for (int nj = 0; nj < 4; ++nj)
      __builtin_prefetch(brow[nj] + k0 + 1024, 0, 1);
  }

  #pragma unroll
  for (int nj = 0; nj < 4; ++nj) {
    const int col = nbase + nj * 16 + r16;
    if (col < NH) {
      const float bias = fb1[col];
      #pragma unroll
      for (int mi = 0; mi < 4; ++mi) {
        #pragma unroll
        for (int r = 0; r < 8; ++r) {
          const int row = mbase + mi * 16 + hi * 8 + r;
          float v = acc[mi][nj][r] + bias;
          H[(size_t)row * NHP + col] = (_Float16)(v > 0.f ? v : 0.f);
        }
      }
    }
  }
}

// ---------------- GEMM2: out = H @ W2p^T + fb2 (W2p is zero-padded 16 x NHP) ----------------
__global__ __launch_bounds__(256) void gemm2_kernel(
    const _Float16* __restrict__ H, const _Float16* __restrict__ W2,
    const float* __restrict__ fb2, float* __restrict__ out) {
  const int wave = threadIdx.x >> 5, lane = threadIdx.x & 31;
  const int r16 = lane & 15, hi = lane >> 4;
  const int mtile = (blockIdx.x * 8 + wave) * 16;
  const _Float16* arow = H  + (size_t)(mtile + r16) * NHP + hi * 8;
  const _Float16* brow = W2 + (size_t)r16 * NHP + hi * 8;
  v8f acc = {};
  for (int k0 = 0; k0 < NHP; k0 += 32) {
    v8h alo = *(const v8h*)(arow + k0);
    v8h ahi = *(const v8h*)(arow + k0 + 16);
    v8h blo = *(const v8h*)(brow + k0);
    v8h bhi = *(const v8h*)(brow + k0 + 16);
    v16h a  = __builtin_shufflevector(alo, ahi, 0,1,2,3,4,5,6,7,8,9,10,11,12,13,14,15);
    v16h bm = __builtin_shufflevector(blo, bhi, 0,1,2,3,4,5,6,7,8,9,10,11,12,13,14,15);
    acc = __builtin_amdgcn_wmma_f32_16x16x32_f16(false, a, false, bm, (short)0, acc, false, false);
  }
  if (r16 < NOUT) {
    const float bias = fb2[r16];
    #pragma unroll
    for (int r = 0; r < 8; ++r) {
      const int row = mtile + hi * 8 + r;
      out[(size_t)row * NOUT + r16] = acc[r] + bias;
    }
  }
}

extern "C" void kernel_launch(void* const* d_in, const int* in_sizes, int n_in,
                              void* d_out, int out_size, void* d_ws, size_t ws_size,
                              hipStream_t stream) {
  (void)in_sizes; (void)n_in; (void)out_size; (void)ws_size;
  const float* x   = (const float*)d_in[0];
  const float* hw1 = (const float*)d_in[1];
  const float* hb1 = (const float*)d_in[2];
  const float* hw2 = (const float*)d_in[3];
  const float* hb2 = (const float*)d_in[4];
  const float* hw3 = (const float*)d_in[5];
  const float* hb3 = (const float*)d_in[6];
  const float* lw1 = (const float*)d_in[7];
  const float* lb1 = (const float*)d_in[8];
  const float* lw2 = (const float*)d_in[9];
  const float* lb2 = (const float*)d_in[10];
  const float* lw3 = (const float*)d_in[11];
  const float* lb3 = (const float*)d_in[12];
  const float* fw1 = (const float*)d_in[13];
  const float* fb1 = (const float*)d_in[14];
  const float* fw2 = (const float*)d_in[15];
  const float* fb2 = (const float*)d_in[16];

  // workspace layout (all 16B aligned; all row strides are multiples of 16B)
  _Float16* F   = (_Float16*)d_ws;              // NB   x NFEATP
  _Float16* W1h = F   + (size_t)NB * NFEATP;    // NH   x NFEATP (zero K-pad)
  _Float16* W2h = W1h + (size_t)NH * NFEATP;    // 16   x NHP    (zero pads)
  _Float16* H   = W2h + (size_t)16 * NHP;       // NB   x NHP

  cvt_w1<<<(NH * NFEATP + 255) / 256, 256, 0, stream>>>(fw1, W1h);
  cvt_w2<<<(16 * NHP + 255) / 256, 256, 0, stream>>>(fw2, W2h);

  features_kernel<<<(NB * NC) / 4, 128, 0, stream>>>(
      x, hw1, hb1, hw2, hb2, hw3, hb3, lw1, lb1, lw2, lb2, lw3, lb3, F);

  gemm1_kernel<<<dim3((NH + 63) / 64, (NB / 64) / 8), 256, 0, stream>>>(F, W1h, fb1, H);
  zero_hpad<<<(NB * 16 + 255) / 256, 256, 0, stream>>>(H);
  gemm2_kernel<<<(NB / 16) / 8, 256, 0, stream>>>(H, W2h, fb2, (float*)d_out);
}
